// ATLASModel_31533649888025
// MI455X (gfx1250) — matrix operand order repaired and checked by reading, not compiled, not run on hardware
//
#include <hip/hip_runtime.h>
#include <hip/hip_bf16.h>
#include <math.h>

// ============================================================================
// ATLAS forward pass for MI455X (gfx1250, wave32, WMMA).
//
// Shapes: B=8 L=1024 NF=32 TOK=56 D=256 H=8 DH=32 DIN=512 DS=16 DTR=16
//         DCONV=4 NL=4 M=2048 AD=5.  BL = B*L = 8192 rows everywhere.
//
// All dense matmuls run on v_wmma_f32_16x16x32_f16 (fp32 staged to LDS as
// f16, fp32 accumulate).  Fragment gathers are fully vectorized:
//  - A fragments: two contiguous 8-half runs per lane -> 2x ds_load_b128
//  - B fragments: operand stored N-major in LDS so each lane's 16 K-values
//    are contiguous -> 2x ds_load_b128  (no more ds_load_u16 storms)
// Self/cross attention are flash-style with online softmax; P tiles overlay
// the score tile (each softmax thread owns its row).  The Mamba scan runs as
// 4096 independent register-resident scans.
//
// Input pointer order assumption: setup_inputs() dict order with the nested
// 'params' pytree flattened JAX-style (dict keys sorted, list in order).
// ============================================================================

typedef __attribute__((ext_vector_type(16))) _Float16 v16h;
typedef __attribute__((ext_vector_type(8)))  _Float16 v8h;
typedef __attribute__((ext_vector_type(2)))  _Float16 v2h;
typedef __attribute__((ext_vector_type(8)))  float    v8f;

// ---------------------------------------------------------------- activations
__device__ __forceinline__ float geluf(float x) {
  float x3 = x * x * x;
  return 0.5f * x * (1.0f + tanhf(0.7978845608f * (x + 0.044715f * x3)));
}
__device__ __forceinline__ float siluf(float x) { return x / (1.0f + expf(-x)); }
__device__ __forceinline__ float softplusf(float x) {
  return (x > 20.0f) ? x : log1pf(expf(x));
}
__device__ __forceinline__ float apply_act(float v, int act) {
  if (act == 1) return geluf(v);
  if (act == 2) return siluf(v);
  if (act == 3) return softplusf(v);
  return v;
}

// ------------------------------------------------------------- WMMA fragments
__device__ __forceinline__ v16h hcombine(v8h lo, v8h hi) {
  return __builtin_shufflevector(lo, hi, 0, 1, 2, 3, 4, 5, 6, 7,
                                 8, 9, 10, 11, 12, 13, 14, 15);
}
// A fragment from a row-major MxK tile (ISA 7.12.2): lane l holds row
// M=rowbase+(l&15); lanes 0-15 K {0-7,16-23}, lanes 16-31 K {8-15,24-31}.
// Two contiguous 8-half runs -> 2x ds_load_b128.
__device__ __forceinline__ v16h frag_a(const _Float16* S, int ld, int rowbase,
                                       int lane) {
  const _Float16* p = S + (rowbase + (lane & 15)) * ld;
  const int kh = (lane >> 4) << 3;  // 0 or 8
  v8h lo = *(const v8h*)(p + kh);
  v8h hi = *(const v8h*)(p + 16 + kh);
  return hcombine(lo, hi);
}
// B fragment from an N-MAJOR tile S[n][k] (i.e. column-major of the logical
// KxN operand): lane l holds col N=colbase+(l&15); lanes 0-15 K {0-7,16-23},
// lanes 16-31 K {8-15,24-31} (matches frag_a). Two 8-half runs -> 2x ds_load_b128.
__device__ __forceinline__ v16h frag_bt(const _Float16* S, int ld, int colbase,
                                        int lane) {
  const _Float16* p = S + (colbase + (lane & 15)) * ld + ((lane >> 4) << 3);
  return hcombine(*(const v8h*)p, *(const v8h*)(p + 16));
}
__device__ __forceinline__ v8f wmma32(v16h a, v16h b, v8f c) {
  return __builtin_amdgcn_wmma_f32_16x16x32_f16(false, a, false, b, (short)0, c,
                                                false, false);
}

// ============================================================================
// Generic WMMA GEMM: C[M,N] = act(A[M,K](lda) @ W[K,N] + bias) + res
// 64x64 tile per block, 128 threads (4 waves), K chunked by 32 (zero padded).
// W tile stored N-major in LDS (stride 40 halves = 80B: 16B-aligned rows,
// conflict-friendly) so B fragments are contiguous.
// ============================================================================
__global__ __launch_bounds__(128) void k_gemm(
    const float* __restrict__ A, int lda,
    const float* __restrict__ W,
    const float* __restrict__ bias,
    const float* __restrict__ res,
    float* __restrict__ C, int ldc,
    int M, int N, int K, int act) {
  __shared__ __align__(16) _Float16 As[64 * 32];
  __shared__ __align__(16) _Float16 Ws[64 * 40];  // [n][k] n-major, pad 40
  const int t = threadIdx.x;
  const int lane = t & 31;
  const int w = t >> 5;
  const int row0 = blockIdx.x * 64;
  const int col0 = blockIdx.y * 64;
  v8f acc[4] = {};
  for (int k0 = 0; k0 < K; k0 += 32) {
    // A tile 64x32 row-major: paired b32 stores, coalesced global reads
#pragma unroll
    for (int i = 0; i < 8; ++i) {
      int idx = t + i * 128;              // 0..1023 pairs
      int r = idx >> 4, c2 = (idx & 15) * 2;
      int gr = row0 + r;
      int gk = k0 + c2;
      float f0 = (gr < M && gk < K) ? A[(size_t)gr * lda + gk] : 0.0f;
      float f1 = (gr < M && gk + 1 < K) ? A[(size_t)gr * lda + gk + 1] : 0.0f;
      v2h hv = {(_Float16)f0, (_Float16)f1};
      *(v2h*)(&As[r * 32 + c2]) = hv;
    }
    // W tile 32x64 -> n-major LDS (transposed store, coalesced global reads)
#pragma unroll
    for (int i = 0; i < 16; ++i) {
      int idx = t + i * 128;
      int r = idx >> 6, c = idx & 63;     // r = k, c = n
      int gk = k0 + r, gc = col0 + c;
      float v = (gk < K && gc < N) ? W[(size_t)gk * N + gc] : 0.0f;
      Ws[c * 40 + r] = (_Float16)v;
    }
    if (k0 + 32 < K) {                    // global_prefetch_b8 next chunk
      __builtin_prefetch(&A[(size_t)(row0 + (t >> 1)) * lda + k0 + 32], 0, 1);
    }
    __syncthreads();
    v16h af = frag_a(As, 32, w * 16, lane);
#pragma unroll
    for (int j = 0; j < 4; ++j) {
      v16h bf = frag_bt(Ws, 40, j * 16, lane);
      acc[j] = wmma32(af, bf, acc[j]);
    }
    __syncthreads();
  }
  const int colb = lane & 15;
  const int rowadd = (lane >> 4) << 3;
#pragma unroll
  for (int j = 0; j < 4; ++j) {
    int gc = col0 + j * 16 + colb;
    if (gc >= N) continue;
    float bv = bias ? bias[gc] : 0.0f;
#pragma unroll
    for (int r = 0; r < 8; ++r) {
      int gr = row0 + w * 16 + r + rowadd;
      if (gr >= M) continue;
      float v = apply_act(acc[j][r] + bv, act);
      if (res) v += res[(size_t)gr * ldc + gc];
      C[(size_t)gr * ldc + gc] = v;
    }
  }
}

// ============================================================================
// Token assembly: features(32) | time2vec(8) | dow(6)+month(8)+flags(2)
// ============================================================================
__global__ void k_token(const float* __restrict__ feat, const float* __restrict__ ts,
                        const int* __restrict__ dow, const int* __restrict__ mon,
                        const int* __restrict__ opx, const int* __restrict__ qe,
                        const float* __restrict__ t2v_w, const float* __restrict__ t2v_b,
                        const float* __restrict__ dow_emb, const float* __restrict__ mon_emb,
                        float* __restrict__ tok) {
  int bl = blockIdx.x * 256 + threadIdx.x;
  if (bl >= 8192) return;
  float* o = tok + (size_t)bl * 56;
  const float* f = feat + (size_t)bl * 32;
#pragma unroll
  for (int i = 0; i < 32; ++i) o[i] = f[i];
  float tv = ts[bl];
  o[32] = tv * t2v_w[0] + t2v_b[0];
#pragma unroll
  for (int j = 1; j < 8; ++j) o[32 + j] = sinf(tv * t2v_w[j] + t2v_b[j]);
  int d = dow[bl], m = mon[bl];
#pragma unroll
  for (int j = 0; j < 6; ++j) o[40 + j] = dow_emb[d * 6 + j];
#pragma unroll
  for (int j = 0; j < 8; ++j) o[46 + j] = mon_emb[m * 8 + j];
  o[54] = (float)opx[bl];
  o[55] = (float)qe[bl];
}

// ============================================================================
// VSN gate: w = softmax(s, 56); tw = token * w.  One wave per row.
// ============================================================================
__global__ __launch_bounds__(256) void k_vsngate(const float* __restrict__ s,
                                                 const float* __restrict__ tok,
                                                 float* __restrict__ tw) {
  int t = threadIdx.x, lane = t & 31, w = t >> 5;
  int row = blockIdx.x * 8 + w;
  const float* sr = s + (size_t)row * 56;
  float v1 = sr[lane];
  bool has2 = (lane + 32) < 56;
  float v2 = has2 ? sr[lane + 32] : -3.0e38f;
  float mx = fmaxf(v1, v2);
#pragma unroll
  for (int m = 16; m >= 1; m >>= 1) mx = fmaxf(mx, __shfl_xor(mx, m, 32));
  float e1 = expf(v1 - mx);
  float e2 = has2 ? expf(v2 - mx) : 0.0f;
  float sum = e1 + e2;
#pragma unroll
  for (int m = 16; m >= 1; m >>= 1) sum += __shfl_xor(sum, m, 32);
  float inv = 1.0f / sum;
  const float* tr = tok + (size_t)row * 56;
  float* o = tw + (size_t)row * 56;
  o[lane] = tr[lane] * e1 * inv;
  if (has2) o[lane + 32] = tr[lane + 32] * e2 * inv;
}

// ============================================================================
// RMSNorm over D=256.  One wave per row.
// ============================================================================
__global__ __launch_bounds__(256) void k_rms(const float* __restrict__ x,
                                             const float* __restrict__ nw,
                                             float* __restrict__ xn) {
  int t = threadIdx.x, lane = t & 31, w = t >> 5;
  int row = blockIdx.x * 8 + w;
  const float* xr = x + (size_t)row * 256;
  float ss = 0.0f;
#pragma unroll
  for (int i = 0; i < 8; ++i) { float v = xr[lane + 32 * i]; ss += v * v; }
#pragma unroll
  for (int m = 16; m >= 1; m >>= 1) ss += __shfl_xor(ss, m, 32);
  float sc = rsqrtf(ss * (1.0f / 256.0f) + 1e-6f);
#pragma unroll
  for (int i = 0; i < 8; ++i) {
    int c = lane + 32 * i;
    xn[(size_t)row * 256 + c] = xr[c] * sc * nw[c];
  }
}

// ============================================================================
// Causal depthwise conv (DCONV=4) + SiLU.  x1 = inproj cols [0,512), ld=1024.
// ============================================================================
__global__ void k_conv(const float* __restrict__ inproj, const float* __restrict__ cw,
                       const float* __restrict__ cb, float* __restrict__ out) {
  int idx = blockIdx.x * 256 + threadIdx.x;
  if (idx >= 8192 * 512) return;
  int row = idx >> 9, c = idx & 511;
  int b = row >> 10, l = row & 1023;
  float acc = cb[c];
#pragma unroll
  for (int kk = 0; kk < 4; ++kk) {
    int lp = l + kk - 3;
    if (lp >= 0) acc += inproj[((size_t)(b * 1024 + lp)) * 1024 + c] * cw[c * 4 + kk];
  }
  out[(size_t)row * 512 + c] = siluf(acc);
}

// ============================================================================
// Mamba selective scan.  One thread per (b, channel): 4096 scans, L=1024.
// ============================================================================
__global__ __launch_bounds__(128) void k_scan(const float* __restrict__ delta,
                                              const float* __restrict__ u,
                                              const float* __restrict__ xw,
                                              const float* __restrict__ A_log,
                                              const float* __restrict__ Dp,
                                              float* __restrict__ ys) {
  int id = blockIdx.x * 128 + threadIdx.x;   // 0..4095
  int b = id >> 9, c = id & 511;
  float A[16];
#pragma unroll
  for (int s = 0; s < 16; ++s) A[s] = -expf(A_log[c * 16 + s]);
  float dp = Dp[c];
  float h[16] = {};
  for (int l = 0; l < 1024; ++l) {
    size_t row = (size_t)(b * 1024 + l);
    float d  = delta[row * 512 + c];
    float uu = u[row * 512 + c];
    const float* xr = xw + row * 48;
    float du = d * uu;
    float acc = 0.0f;
#pragma unroll
    for (int s = 0; s < 16; ++s) {
      float e = __expf(d * A[s]);
      h[s] = e * h[s] + du * xr[16 + s];
      acc += h[s] * xr[32 + s];
    }
    ys[row * 512 + c] = acc + uu * dp;
  }
}

// y *= silu(z); z = inproj cols [512,1024)
__global__ void k_gate(float* __restrict__ y, const float* __restrict__ inproj) {
  int idx = blockIdx.x * 256 + threadIdx.x;
  if (idx >= 8192 * 512) return;
  int row = idx >> 9, c = idx & 511;
  y[idx] *= siluf(inproj[(size_t)row * 1024 + 512 + c]);
}

// ============================================================================
// sig[b,l] = mean(log sigma); pooled per b -> tau MLP (1->32->1, exp)
// ============================================================================
__global__ __launch_bounds__(256) void k_sigtau(const float* __restrict__ sigma,
                                                const float* __restrict__ tw1,
                                                const float* __restrict__ tb1,
                                                const float* __restrict__ tw2,
                                                const float* __restrict__ tb2,
                                                float* __restrict__ sig,
                                                float* __restrict__ tau) {
  __shared__ float red[256];
  int b = blockIdx.x, t = threadIdx.x;
  float local = 0.0f;
  for (int l = t; l < 1024; l += 256) {
    const float* sp = sigma + ((size_t)(b * 1024 + l)) * 32;
    float s = 0.0f;
#pragma unroll
    for (int f = 0; f < 32; ++f) s += logf(sp[f]);
    s *= (1.0f / 32.0f);
    sig[b * 1024 + l] = s;
    local += s;
  }
  red[t] = local;
  __syncthreads();
  for (int m = 128; m >= 1; m >>= 1) {
    if (t < m) red[t] += red[t + m];
    __syncthreads();
  }
  if (t == 0) {
    float pooled = red[0] * (1.0f / 1024.0f);
    float acc = tb2[0];
#pragma unroll
    for (int j = 0; j < 32; ++j) acc += geluf(pooled * tw1[j] + tb1[j]) * tw2[j];
    tau[b] = expf(acc);
  }
}

// dsh[b,l] = MLP(sig[b,l])  (1->32->1)
__global__ void k_dsh(const float* __restrict__ sig, const float* __restrict__ dw1,
                      const float* __restrict__ db1, const float* __restrict__ dw2,
                      const float* __restrict__ db2, float* __restrict__ dsh) {
  int bl = blockIdx.x * 256 + threadIdx.x;
  if (bl >= 8192) return;
  float s = sig[bl];
  float acc = db2[0];
#pragma unroll
  for (int j = 0; j < 32; ++j) acc += geluf(s * dw1[j] + db1[j]) * dw2[j];
  dsh[bl] = acc;
}

// ============================================================================
// De-stationary causal flash self-attention. grid (L/32, B*H), 64 threads.
// q pre-scaled by tau[b]/sqrt(DH); dsh added per key; causal mask; online
// softmax; QK^T and PV both on WMMA.  P overlays Sld (each softmax thread
// owns its full row).  V tile stored feature-major for vector B fragments.
// ============================================================================
__global__ __launch_bounds__(64) void k_attn(const float* __restrict__ q,
                                             const float* __restrict__ k,
                                             const float* __restrict__ v,
                                             const float* __restrict__ tau,
                                             const float* __restrict__ dsh,
                                             float* __restrict__ out) {
  __shared__ __align__(16) _Float16 qs[32 * 32];   // row-major [q][dh]
  __shared__ __align__(16) _Float16 ks[32 * 32];   // [key][dh] (N-major for S)
  __shared__ __align__(16) _Float16 vt[32 * 40];   // [dh][key] (N-major for PV)
  __shared__ __align__(16) float    Sld[32 * 32];  // scores; P overlays rows
  __shared__ float mrow[32], lrow[32], arow[32];
  _Float16* Ph = (_Float16*)Sld;                   // row stride 64 halves
  const int t = threadIdx.x, lane = t & 31, w = t >> 5;
  const int bh = blockIdx.y, b = bh >> 3, h = bh & 7;
  const int qt = blockIdx.x, qbase = qt * 32;
  const float qscale = tau[b] * 0.1767766953f;     // tau / sqrt(32)
#pragma unroll
  for (int i = 0; i < 8; ++i) {                    // q tile, paired stores
    int idx = t + i * 64;                          // 0..511 pairs
    int r = idx >> 4, c2 = (idx & 15) * 2;
    size_t g = ((size_t)(b * 1024 + qbase + r)) * 256 + h * 32 + c2;
    v2h hv = {(_Float16)(q[g] * qscale), (_Float16)(q[g + 1] * qscale)};
    *(v2h*)(&qs[r * 32 + c2]) = hv;
  }
  if (t < 32) { mrow[t] = -3.0e38f; lrow[t] = 0.0f; }
  v8f o0 = {}, o1 = {};
  __syncthreads();
  for (int kt = 0; kt <= qt; ++kt) {
    int kb = kt * 32;
#pragma unroll
    for (int i = 0; i < 8; ++i) {                  // k tile, paired stores
      int idx = t + i * 64;
      int r = idx >> 4, c2 = (idx & 15) * 2;
      size_t g = ((size_t)(b * 1024 + kb + r)) * 256 + h * 32 + c2;
      v2h hv = {(_Float16)k[g], (_Float16)k[g + 1]};
      *(v2h*)(&ks[r * 32 + c2]) = hv;
    }
    for (int i = t; i < 32 * 32; i += 64) {        // v tile, transposed store
      int r = i >> 5, c = i & 31;                  // r = key, c = dh
      vt[c * 40 + r] = (_Float16)v[((size_t)(b * 1024 + kb + r)) * 256 + h * 32 + c];
    }
    __syncthreads();
    v16h af = frag_a(qs, 32, w * 16, lane);
#pragma unroll
    for (int j = 0; j < 2; ++j) {
      v8f s = {};
      v16h bf = frag_bt(ks, 32, j * 16, lane);
      s = wmma32(af, bf, s);
      int ci = j * 16 + (lane & 15);
      int gk = kb + ci;
      float db = dsh[b * 1024 + gk];
#pragma unroll
      for (int r = 0; r < 8; ++r) {
        int ri = w * 16 + r + ((lane >> 4) << 3);
        float val = s[r] + db;
        if (gk > qbase + ri) val = -1.0e9f;        // causal (after dsh, like ref)
        Sld[ri * 32 + ci] = val;
      }
    }
    __syncthreads();
    if (t < 32) {                                  // softmax; P overlays S row
      float mx = -3.0e38f;
      float srow[32];
#pragma unroll
      for (int c = 0; c < 32; ++c) { srow[c] = Sld[t * 32 + c]; mx = fmaxf(mx, srow[c]); }
      float newm = fmaxf(mrow[t], mx);
      float al = expf(mrow[t] - newm);
      float sum = 0.0f;
#pragma unroll
      for (int c = 0; c < 32; ++c) {
        float p = expf(srow[c] - newm);
        Ph[t * 64 + c] = (_Float16)p;
        sum += p;
      }
      lrow[t] = lrow[t] * al + sum;
      mrow[t] = newm;
      arow[t] = al;
    }
    __syncthreads();
    {
      int rowadd = (lane >> 4) << 3;
#pragma unroll
      for (int r = 0; r < 8; ++r) {
        float al = arow[w * 16 + r + rowadd];
        o0[r] *= al; o1[r] *= al;
      }
      v16h ap = frag_a(Ph, 64, w * 16, lane);
      v16h b0 = frag_bt(vt, 40, 0, lane);
      v16h b1 = frag_bt(vt, 40, 16, lane);
      o0 = wmma32(ap, b0, o0);
      o1 = wmma32(ap, b1, o1);
    }
    __syncthreads();
  }
  int rowadd = (lane >> 4) << 3, cb = lane & 15;
#pragma unroll
  for (int r = 0; r < 8; ++r) {
    int ri = w * 16 + r + rowadd;
    float inv = 1.0f / lrow[ri];
    size_t g = ((size_t)(b * 1024 + qbase + ri)) * 256 + h * 32;
    out[g + cb]      = o0[r] * inv;
    out[g + 16 + cb] = o1[r] * inv;
  }
}

// ============================================================================
// Flash cross-attention over memory bank (M=2048, full D=256 dot products).
// grid BL/32, 128 threads (4 waves).  One LDS buffer holds the key tile
// (key-major) then the value tile (feature-major, padded stride 72).
// P overlays Sld.  ~62KB LDS total.
// ============================================================================
__global__ __launch_bounds__(128) void k_cross(const float* __restrict__ cq,
                                               const float* __restrict__ ck,
                                               const float* __restrict__ cv,
                                               float* __restrict__ out) {
  __shared__ __align__(16) _Float16 qs[32 * 256];   // 16KB [q][feat]
  __shared__ __align__(16) _Float16 kvb[256 * 72];  // 36KB keys[64][256] / vt[256][72]
  __shared__ __align__(16) float    Sld[32 * 64];   // 8KB; P overlays (stride 128h)
  __shared__ float mrow[32], lrow[32], arow[32];
  _Float16* Ph = (_Float16*)Sld;
  const int t = threadIdx.x, lane = t & 31, w = t >> 5;
  const int qbase = blockIdx.x * 32;
  const int wr = (w >> 1) * 16;        // wave's row base (0|16)
  const int wc = (w & 1) * 128;        // wave's O column base (0|128)
#pragma unroll
  for (int i = 0; i < 32; ++i) {       // q tile, paired stores
    int idx = t + i * 128;             // 0..4095 pairs
    int r = idx >> 7, c2 = (idx & 127) * 2;
    size_t g = (size_t)(qbase + r) * 256 + c2;
    v2h hv = {(_Float16)(cq[g] * 0.0625f), (_Float16)(cq[g + 1] * 0.0625f)};
    *(v2h*)(&qs[r * 256 + c2]) = hv;
  }
  if (t < 32) { mrow[t] = -3.0e38f; lrow[t] = 0.0f; }
  v8f o[8] = {};
  __syncthreads();
  for (int mt = 0; mt < 32; ++mt) {
    int mb = mt * 64;
#pragma unroll
    for (int i = 0; i < 64; ++i) {     // key tile [key][feat], paired stores
      int idx = t + i * 128;           // 0..8191 pairs
      int r = idx >> 7, c2 = (idx & 127) * 2;
      size_t g = (size_t)(mb + r) * 256 + c2;
      v2h hv = {(_Float16)ck[g], (_Float16)ck[g + 1]};
      *(v2h*)(&kvb[r * 256 + c2]) = hv;
    }
    __syncthreads();
#pragma unroll
    for (int j = 0; j < 2; ++j) {      // S rows wr.., cols (w&1)*32+16j
      v8f s = {};
      int cb = (w & 1) * 32 + j * 16;
#pragma unroll
      for (int kc = 0; kc < 8; ++kc) { // K = 256 chained
        v16h af = frag_a(qs + kc * 32, 256, wr, lane);
        v16h bf = frag_bt(kvb + kc * 32, 256, cb, lane);
        s = wmma32(af, bf, s);
      }
      int ci = cb + (lane & 15);
#pragma unroll
      for (int r = 0; r < 8; ++r) {
        int ri = wr + r + ((lane >> 4) << 3);
        Sld[ri * 64 + ci] = s[r];
      }
    }
    __syncthreads();
    if (t < 32) {                      // softmax; P overlays S row (owned)
      float mx = -3.0e38f;
      float srow[64];
#pragma unroll
      for (int c = 0; c < 64; ++c) { srow[c] = Sld[t * 64 + c]; mx = fmaxf(mx, srow[c]); }
      float newm = fmaxf(mrow[t], mx);
      float al = expf(mrow[t] - newm);
      float sum = 0.0f;
#pragma unroll
      for (int c = 0; c < 64; ++c) {
        float p = expf(srow[c] - newm);
        Ph[t * 128 + c] = (_Float16)p;
        sum += p;
      }
      lrow[t] = lrow[t] * al + sum;
      mrow[t] = newm;
      arow[t] = al;
    }
    __syncthreads();
    for (int i = t; i < 64 * 256; i += 128) {  // value tile -> feature-major
      int r = i >> 8, c = i & 255;             // r = key, c = feat
      kvb[c * 72 + r] = (_Float16)cv[(size_t)(mb + r) * 256 + c];
    }
    __syncthreads();
    {
      int rowadd = (lane >> 4) << 3;
      float al[8];
#pragma unroll
      for (int r = 0; r < 8; ++r) al[r] = arow[wr + r + rowadd];
#pragma unroll
      for (int j = 0; j < 8; ++j)
#pragma unroll
        for (int r = 0; r < 8; ++r) o[j][r] *= al[r];
      v16h ap0 = frag_a(Ph, 128, wr, lane);        // keys 0..31
      v16h ap1 = frag_a(Ph + 32, 128, wr, lane);   // keys 32..63
#pragma unroll
      for (int j = 0; j < 8; ++j) {
        int cb = wc + j * 16;
        v16h b0 = frag_bt(kvb, 72, cb, lane);        // keys 0..31
        v16h b1 = frag_bt(kvb + 32, 72, cb, lane);   // keys 32..63
        o[j] = wmma32(ap0, b0, o[j]);
        o[j] = wmma32(ap1, b1, o[j]);
      }
    }
    __syncthreads();
  }
  int rowadd = (lane >> 4) << 3, cb15 = lane & 15;
#pragma unroll
  for (int r = 0; r < 8; ++r) {
    int ri = wr + r + rowadd;
    float inv = 1.0f / lrow[ri];
    size_t g = (size_t)(qbase + ri) * 256;
#pragma unroll
    for (int j = 0; j < 8; ++j) out[g + wc + j * 16 + cb15] = o[j][r] * inv;
  }
}

// ============================================================================
// Final head: r=gelu(rtg*W+b); fused=gelu(cat(self,cross,r)@fus_w+b);
// out = tanh(gelu(fused@ah_w1+b)@ah_w2+b).  One block per batch.
// ============================================================================
__global__ __launch_bounds__(256) void k_head(
    const float* __restrict__ self_out, const float* __restrict__ cross_out,
    const float* __restrict__ rtg, const float* __restrict__ rtg_w,
    const float* __restrict__ rtg_b, const float* __restrict__ fus_w,
    const float* __restrict__ fus_b, const float* __restrict__ ah_w1,
    const float* __restrict__ ah_b1, const float* __restrict__ ah_w2,
    const float* __restrict__ ah_b2, float* __restrict__ out) {
  __shared__ float rr[256], fused[256], hid[128];
  int b = blockIdx.x, t = threadIdx.x;
  rr[t] = geluf(rtg[b] * rtg_w[t] + rtg_b[t]);
  __syncthreads();
  const float* sl = self_out + (size_t)(b * 1024 + 1023) * 256;
  const float* cl = cross_out + (size_t)(b * 1024 + 1023) * 256;
  float acc = fus_b[t];
  for (int i = 0; i < 256; ++i) acc += sl[i] * fus_w[i * 256 + t];
  for (int i = 0; i < 256; ++i) acc += cl[i] * fus_w[(256 + i) * 256 + t];
  for (int i = 0; i < 256; ++i) acc += rr[i] * fus_w[(512 + i) * 256 + t];
  fused[t] = geluf(acc);
  __syncthreads();
  if (t < 128) {
    float a = ah_b1[t];
    for (int i = 0; i < 256; ++i) a += fused[i] * ah_w1[i * 128 + t];
    hid[t] = geluf(a);
  }
  __syncthreads();
  if (t < 5) {
    float a = ah_b2[t];
    for (int i = 0; i < 128; ++i) a += hid[i] * ah_w2[i * 5 + t];
    out[b * 5 + t] = tanhf(a);
  }
}

// ============================================================================
// Host orchestration
// ============================================================================
// Param flattening assumption (JAX pytree: sorted dict keys, list in order):
enum {
  IN_FEAT = 0, IN_TS, IN_DOW, IN_MON, IN_OPX, IN_QE, IN_MU, IN_SIGMA, IN_RTG,
  P_AH_B1 = 9, P_AH_B2, P_AH_W1, P_AH_W2,
  P_CA_WK = 13, P_CA_WO, P_CA_WQ, P_CA_WV,
  P_DEL_B1 = 17, P_DEL_B2, P_DEL_W1, P_DEL_W2,
  P_DOW_EMB = 21, P_FUS_B, P_FUS_W,
  P_MAMBA0 = 24,  // per layer (sorted): A_log, Dp, b_dt, conv_b, conv_w,
                  //                     norm_w, w_dt, w_in, w_out, w_x
  P_MEM_K = 64, P_MEM_V, P_MON_EMB, P_RTG_B, P_RTG_W,
  P_SA_WK = 69, P_SA_WO, P_SA_WQ, P_SA_WV,
  P_T2V_B = 73, P_T2V_W,
  P_TAU_B1 = 75, P_TAU_B2, P_TAU_W1, P_TAU_W2,
  P_VSN_B1 = 79, P_VSN_B2, P_VSN_BV, P_VSN_W1, P_VSN_W2, P_VSN_WV
};

extern "C" void kernel_launch(void* const* d_in, const int* in_sizes, int n_in,
                              void* d_out, int out_size, void* d_ws, size_t ws_size,
                              hipStream_t stream) {
  (void)in_sizes; (void)n_in; (void)out_size; (void)ws_size;
  auto F = [&](int i) { return (const float*)d_in[i]; };
  auto I = [&](int i) { return (const int*)d_in[i]; };
  float* out = (float*)d_out;

  // ---- workspace layout (floats) -------------------------------------------
  float* ws = (float*)d_ws;
  size_t o = 0;
  float* tok    = ws + o; o += (size_t)8192 * 56;
  float* x      = ws + o; o += (size_t)8192 * 256;
  float* xn     = ws + o; o += (size_t)8192 * 256;   // also vsn hidden, cq
  float* inproj = ws + o; o += (size_t)8192 * 1024;  // also q/k/v, ck/cv/cross
  float* convb  = ws + o; o += (size_t)8192 * 512;   // also vsn tw, sa
  float* xw     = ws + o; o += (size_t)8192 * 48;
  float* delta  = ws + o; o += (size_t)8192 * 512;   // also vsn s, self_out
  float* ys     = ws + o; o += (size_t)8192 * 512;
  float* sig    = ws + o; o += 8192;
  float* dshb   = ws + o; o += 8192;
  float* taub   = ws + o; o += 8;

  float* qb = inproj;
  float* kb = inproj + (size_t)8192 * 256;
  float* vb = inproj + (size_t)2 * 8192 * 256;
  float* ckb = inproj;
  float* cvb = inproj + (size_t)2048 * 256;
  float* crossraw = inproj + (size_t)2 * 2048 * 256;
  float* crossout = crossraw + (size_t)8192 * 256;
  float* selfout = delta;

  auto gemm = [&](const float* A, int lda, const float* W, const float* bias,
                  const float* res, float* C, int ldc, int M, int N, int K, int act) {
    dim3 g((M + 63) / 64, (N + 63) / 64);
    k_gemm<<<g, 128, 0, stream>>>(A, lda, W, bias, res, C, ldc, M, N, K, act);
  };
  const int BL = 8192;

  // ---- tokens + VSN --------------------------------------------------------
  k_token<<<32, 256, 0, stream>>>(F(IN_FEAT), F(IN_TS), I(IN_DOW), I(IN_MON),
                                  I(IN_OPX), I(IN_QE), F(P_T2V_W), F(P_T2V_B),
                                  F(P_DOW_EMB), F(P_MON_EMB), tok);
  gemm(tok, 56, F(P_VSN_W1), F(P_VSN_B1), nullptr, xn, 256, BL, 256, 56, 1);   // gelu
  gemm(xn, 256, F(P_VSN_W2), F(P_VSN_B2), nullptr, delta, 56, BL, 56, 256, 0);
  k_vsngate<<<1024, 256, 0, stream>>>(delta, tok, convb);
  gemm(convb, 56, F(P_VSN_WV), F(P_VSN_BV), nullptr, x, 256, BL, 256, 56, 0);

  // ---- Mamba layers --------------------------------------------------------
  for (int li = 0; li < 4; ++li) {
    int pb = P_MAMBA0 + 10 * li;
    const float* A_log  = F(pb + 0);
    const float* Dp     = F(pb + 1);
    const float* b_dt   = F(pb + 2);
    const float* conv_b = F(pb + 3);
    const float* conv_w = F(pb + 4);
    const float* norm_w = F(pb + 5);
    const float* w_dt   = F(pb + 6);
    const float* w_in   = F(pb + 7);
    const float* w_out  = F(pb + 8);
    const float* w_x    = F(pb + 9);
    k_rms<<<1024, 256, 0, stream>>>(x, norm_w, xn);
    gemm(xn, 256, w_in, nullptr, nullptr, inproj, 1024, BL, 1024, 256, 0);
    k_conv<<<16384, 256, 0, stream>>>(inproj, conv_w, conv_b, convb);
    gemm(convb, 512, w_x, nullptr, nullptr, xw, 48, BL, 48, 512, 0);
    gemm(xw, 48, w_dt, b_dt, nullptr, delta, 512, BL, 512, 16, 3);             // softplus
    k_scan<<<32, 128, 0, stream>>>(delta, convb, xw, A_log, Dp, ys);
    k_gate<<<16384, 256, 0, stream>>>(ys, inproj);
    gemm(ys, 512, w_out, nullptr, x, x, 256, BL, 256, 512, 0);                 // residual
  }

  // ---- de-stationary factors ----------------------------------------------
  k_sigtau<<<8, 256, 0, stream>>>(F(IN_SIGMA), F(P_TAU_W1), F(P_TAU_B1),
                                  F(P_TAU_W2), F(P_TAU_B2), sig, taub);
  k_dsh<<<32, 256, 0, stream>>>(sig, F(P_DEL_W1), F(P_DEL_B1), F(P_DEL_W2),
                                F(P_DEL_B2), dshb);

  // ---- causal self-attention ----------------------------------------------
  gemm(x, 256, F(P_SA_WQ), nullptr, nullptr, qb, 256, BL, 256, 256, 0);
  gemm(x, 256, F(P_SA_WK), nullptr, nullptr, kb, 256, BL, 256, 256, 0);
  gemm(x, 256, F(P_SA_WV), nullptr, nullptr, vb, 256, BL, 256, 256, 0);
  k_attn<<<dim3(32, 64), 64, 0, stream>>>(qb, kb, vb, taub, dshb, convb);
  gemm(convb, 256, F(P_SA_WO), nullptr, x, selfout, 256, BL, 256, 256, 0);

  // ---- cross-attention over memory bank -----------------------------------
  gemm(F(P_MEM_K), 256, F(P_CA_WK), nullptr, nullptr, ckb, 256, 2048, 256, 256, 0);
  gemm(F(P_MEM_V), 256, F(P_CA_WV), nullptr, nullptr, cvb, 256, 2048, 256, 256, 0);
  gemm(selfout, 256, F(P_CA_WQ), nullptr, nullptr, xn, 256, BL, 256, 256, 0);
  k_cross<<<256, 128, 0, stream>>>(xn, ckb, cvb, crossraw);
  gemm(crossraw, 256, F(P_CA_WO), nullptr, nullptr, crossout, 256, BL, 256, 256, 0);

  // ---- fusion + action head ------------------------------------------------
  k_head<<<8, 256, 0, stream>>>(selfout, crossout, F(IN_RTG), F(P_RTG_W),
                                F(P_RTG_B), F(P_FUS_W), F(P_FUS_B), F(P_AH_W1),
                                F(P_AH_B1), F(P_AH_W2), F(P_AH_B2), out);
}
